// CharRNN_68152541053688
// MI455X (gfx1250) — matrix operand order repaired and checked by reading, not compiled, not run on hardware
//
#include <hip/hip_runtime.h>
#include <hip/hip_bf16.h>

// ---------------------------------------------------------------------------
// CDNA5 (gfx1250) GRU char-RNN forward cost.  V=8000 B=64 T=256 U=128 L=2.
// Fragment-major WMMA operands (2x b128 per frag); two-phase dual-accumulator
// gates GEMM (A loaded once -> two WMMAs, <=150 live VGPRs); pipelined B in
// the logits GEMM; float4 LDS scans (rolled loops); TDM-staged x_t.
// ---------------------------------------------------------------------------

typedef __attribute__((ext_vector_type(16))) _Float16 v16h;
typedef __attribute__((ext_vector_type(8)))  float    v8f;
typedef __attribute__((ext_vector_type(4)))  unsigned int v4u;
typedef __attribute__((ext_vector_type(8)))  int      v8i;
typedef __attribute__((ext_vector_type(4)))  int      v4i;

#define B_  64
#define T_  256
#define U_  128
#define L_  2
#define V_  8000
#define VP_ 8192   // padded vocab (64 strips of 128)

#if defined(__AMDGCN__) && __has_builtin(__builtin_amdgcn_tensor_load_to_lds) && \
    __has_builtin(__builtin_amdgcn_s_wait_tensorcnt)
#define HAVE_TDM 1
#else
#define HAVE_TDM 0
#endif

// ---------------- WMMA helpers ----------------

__device__ __forceinline__ v8f wmma_f16(v16h a, v16h b, v8f c) {
  return __builtin_amdgcn_wmma_f32_16x16x32_f16(false, a, false, b, (short)0, c,
                                                false, false);
}

__device__ __forceinline__ v16h load_frag(const _Float16* base) {
  return *(const v16h*)(base + (threadIdx.x & 31) * 16);
}

// element (r,c) -> swizzled A storage index (16x32 tiles, (r>>4)*KT + (c>>5))
__device__ __forceinline__ int a_swz64(int r, int c, int KT) {
  const int m    = r & 15;
  const int kp   = c & 31;
  const int rem  = kp & 15;
  const int lane = m + ((rem & 8) ? 16 : 0);
  const int v    = ((kp & 16) ? 4 : 0) + ((rem & 7) >> 1);
  const int h    = (v << 1) + (kp & 1);
  const int tile = (r >> 4) * KT + (c >> 5);
  return tile * 512 + lane * 16 + h;
}

// element (k,n) -> swizzled B storage index (32x16 tiles, (n>>4)*KT + (k>>5))
__device__ __forceinline__ int b_swz(int k, int n, int KT) {
  const int lane = (((k >> 3) & 1) << 4) + (n & 15);
  const int h    = ((k & 16) ? 8 : 0) + (k & 7);
  const int tile = (n >> 4) * KT + (k >> 5);
  return tile * 512 + lane * 16 + h;
}

__device__ __forceinline__ void store_frag_c(float* dst, int ld, v8f c) {
  const int lane  = threadIdx.x & 31;
  const int n     = lane & 15;
  const int mbase = (lane >> 4) << 3;
#pragma unroll
  for (int i = 0; i < 8; ++i) dst[(mbase + i) * ld + n] = c[i];
}

__device__ __forceinline__ v8f zero_c() {
  v8f c;
#pragma unroll
  for (int i = 0; i < 8; ++i) c[i] = 0.0f;
  return c;
}

#if HAVE_TDM
// 1-D contiguous global->LDS DMA via Tensor Data Mover (ISA ch.8 D# layout).
__device__ __forceinline__ void tdm_load_1d(unsigned lds_off,
                                            const void* gptr, unsigned bytes) {
  const unsigned long long ga = (unsigned long long)gptr;
  const unsigned units = bytes >> 3;
  v4u g0;
  g0.x = 1u;                                               // count=1 (valid D#)
  g0.y = lds_off;                                          // LDS byte address
  g0.z = (unsigned)ga;                                     // global_addr[31:0]
  g0.w = (unsigned)((ga >> 32) & 0x1FFFFFFu) | (2u << 30); // addr[56:32]|type=2
  v8i g1;
  g1[0] = (int)(3u << 16);                 // data_size=8B, wg_mask=0, no iter
  g1[1] = (int)((units & 0xFFFFu) << 16);  // tensor_dim0 [15:0]
  g1[2] = (int)(units >> 16);              // tensor_dim0 [31:16], dim1=0
  g1[3] = (int)((units & 0xFFFFu) << 16);  // tile_dim0
  g1[4] = 0; g1[5] = 0; g1[6] = 0; g1[7] = 0;
  v4i zz = {0, 0, 0, 0};
#if __clang_major__ >= 23
  v8i z8 = {0, 0, 0, 0, 0, 0, 0, 0};
  __builtin_amdgcn_tensor_load_to_lds(g0, g1, zz, zz, z8, 0);
#else
  __builtin_amdgcn_tensor_load_to_lds(g0, g1, zz, zz, 0);
#endif
}
#endif

// ---------------- prep kernels ----------------

__global__ void k_swz_w(const float* __restrict__ src,
                        _Float16* __restrict__ dst, int K, int N) {
  int i = blockIdx.x * blockDim.x + threadIdx.x;
  if (i >= K * N) return;
  int k = i / N, n = i % N;
  dst[b_swz(k, n, K >> 5)] = (_Float16)src[i];
}

__global__ void k_swz_sw(const float* __restrict__ sw,
                         _Float16* __restrict__ dst) {
  int i = blockIdx.x * blockDim.x + threadIdx.x;
  if (i >= U_ * VP_) return;
  int k = i / VP_, n = i % VP_;
  float v = (n < V_) ? sw[k * V_ + n] : 0.0f;
  dst[b_swz(k, n, U_ >> 5)] = (_Float16)v;
}

__global__ void k_pad_sb(const float* __restrict__ sb, float* __restrict__ dst) {
  int i = blockIdx.x * blockDim.x + threadIdx.x;
  if (i >= VP_) return;
  dst[i] = (i < V_) ? sb[i] : -1.0e30f;
}

__global__ void k_embed(const float* __restrict__ emb,
                        const int* __restrict__ inputs,
                        _Float16* __restrict__ x) {
  int i = blockIdx.x * blockDim.x + threadIdx.x;      // T_*B_*U_
  if (i >= T_ * B_ * U_) return;
  int t = i / (B_ * U_);
  int r = (i / U_) % B_;
  int u = i % U_;
  int tok = inputs[r * T_ + t];
  x[t * (B_ * U_) + a_swz64(r, u, U_ >> 5)] = (_Float16)emb[tok * U_ + u];
}

// ---------------- persistent GRU kernel (1 workgroup, 8 waves) ----------------
// dyn LDS: inp(16KB @off0) hb(32KB) rh(16KB) zb(16KB) pre(f32 64KB)
//          red(512 f32) par(1536 f32: g_g|b_g|g_c|b_c)
__global__ void __launch_bounds__(256)
k_gru(const _Float16* __restrict__ x,
      const _Float16* __restrict__ Wxg, const _Float16* __restrict__ Whg,
      const _Float16* __restrict__ Wxc, const _Float16* __restrict__ Whc,
      const float* __restrict__ g_g, const float* __restrict__ b_g,
      const float* __restrict__ g_c, const float* __restrict__ b_c,
      _Float16* __restrict__ outh) {
  extern __shared__ __align__(32) char smem[];
  _Float16* inp = (_Float16*)smem;                 // LDS offset 0
  _Float16* hb  = inp + B_ * U_;
  _Float16* rh  = hb + L_ * B_ * U_;
  _Float16* zb  = rh + B_ * U_;
  float*    pre = (float*)(zb + B_ * U_);
  float*    red = pre + B_ * 2 * U_;
  float*    par = red + 512;                       // params cache

  const int tid  = threadIdx.x;
  const int wave = tid >> 5;

  for (int i = tid; i < L_ * B_ * U_; i += 256) hb[i] = (_Float16)0.0f;
  for (int i = tid; i < 512; i += 256) { par[i] = g_g[i]; par[512 + i] = b_g[i]; }
  if (tid < 256) {
    par[1024 + tid] = g_c[tid];
    par[1280 + tid] = b_c[tid];
  }

#if HAVE_TDM
  if (tid < 32) {                       // wave 0 DMAs x_0 into inp (LDS off 0)
    tdm_load_1d(0u, x, B_ * U_ * 2);
    __builtin_amdgcn_s_wait_tensorcnt((short)0);
  }
#endif
  __syncthreads();

  for (int t = 0; t < T_; ++t) {
#if !HAVE_TDM
    {
      const uint4* s = (const uint4*)(x + (size_t)t * B_ * U_);
      uint4* d = (uint4*)inp;
      for (int i = tid; i < (B_ * U_) / 8; i += 256) d[i] = s[i];
    }
    __syncthreads();
#endif

    for (int l = 0; l < L_; ++l) {
      const _Float16* wxg = Wxg + l * U_ * 2 * U_;
      const _Float16* whg = Whg + l * U_ * 2 * U_;
      const _Float16* wxc = Wxc + l * U_ * U_;
      const _Float16* whc = Whc + l * U_ * U_;
      const _Float16* hl  = hb + l * B_ * U_;

      // ---- gates GEMM: pre[64][256] = inp@Wxg + h@Whg ----
      // two phases (x then h); 8 persistent accumulators; B registers reused.
      {
        const int nt0 = wave * 2, nt1 = nt0 + 1;
        v8f acc[8];
#pragma unroll
        for (int i = 0; i < 8; ++i) acc[i] = zero_c();
        {
          v16h b0[4], b1[4];
#pragma unroll
          for (int k = 0; k < 4; ++k) {
            b0[k] = load_frag(wxg + (nt0 * 4 + k) * 512);
            b1[k] = load_frag(wxg + (nt1 * 4 + k) * 512);
          }
#pragma unroll
          for (int mt = 0; mt < 4; ++mt)
#pragma unroll
            for (int k = 0; k < 4; ++k) {
              v16h a = load_frag(inp + (mt * 4 + k) * 512);
              acc[mt * 2]     = wmma_f16(a, b0[k], acc[mt * 2]);
              acc[mt * 2 + 1] = wmma_f16(a, b1[k], acc[mt * 2 + 1]);
            }
        }
        {
          v16h b0[4], b1[4];
#pragma unroll
          for (int k = 0; k < 4; ++k) {
            b0[k] = load_frag(whg + (nt0 * 4 + k) * 512);
            b1[k] = load_frag(whg + (nt1 * 4 + k) * 512);
          }
#pragma unroll
          for (int mt = 0; mt < 4; ++mt)
#pragma unroll
            for (int k = 0; k < 4; ++k) {
              v16h a = load_frag(hl + (mt * 4 + k) * 512);
              acc[mt * 2]     = wmma_f16(a, b0[k], acc[mt * 2]);
              acc[mt * 2 + 1] = wmma_f16(a, b1[k], acc[mt * 2 + 1]);
            }
        }
#pragma unroll
        for (int mt = 0; mt < 4; ++mt) {
          store_frag_c(pre + mt * 16 * 2 * U_ + nt0 * 16, 2 * U_, acc[mt * 2]);
          store_frag_c(pre + mt * 16 * 2 * U_ + nt1 * 16, 2 * U_, acc[mt * 2 + 1]);
        }
      }
      __syncthreads();

      // ---- LN + sigmoid over 256 cols; 4 threads/row; float4 scans ----
      {
        const int q = tid >> 6, r = tid & 63;
        const float4* p4 = (const float4*)(pre + r * 2 * U_ + q * 64);
        float s = 0.0f, s2 = 0.0f;
        for (int j = 0; j < 16; ++j) {
          float4 v = p4[j];
          s  += v.x + v.y + v.z + v.w;
          s2 += v.x * v.x + v.y * v.y + v.z * v.z + v.w * v.w;
        }
        red[q * 64 + r] = s;
        red[256 + q * 64 + r] = s2;
        __syncthreads();
        float mean = 0.0f, msq = 0.0f;
        for (int i = 0; i < 4; ++i) {
          mean += red[i * 64 + r];
          msq  += red[256 + i * 64 + r];
        }
        mean *= (1.0f / 256.0f);
        msq  *= (1.0f / 256.0f);
        const float rs = rsqrtf(msq - mean * mean + 1e-5f);
        const float* gg = par + l * 2 * U_;
        const float* bg = par + 512 + l * 2 * U_;
        auto gate1 = [&](float pv, int c) {
          float val = (pv - mean) * rs * gg[c] + bg[c];
          float sig = 1.0f / (1.0f + __expf(-val));
          if (c < U_) {
            const int si = a_swz64(r, c, 4);
            rh[si] = (_Float16)(sig * (float)hl[si]);
          } else {
            zb[r * U_ + (c - U_)] = (_Float16)sig;
          }
        };
        for (int j = 0; j < 16; ++j) {
          float4 v = p4[j];
          const int c0 = q * 64 + j * 4;
          gate1(v.x, c0);
          gate1(v.y, c0 + 1);
          gate1(v.z, c0 + 2);
          gate1(v.w, c0 + 3);
        }
      }
      __syncthreads();

      // ---- candidate GEMM: pre[64][128] = inp@Wxc + rh@Whc ----
      {
        const int nt = wave;
        v16h bxc[4], bhc[4];
#pragma unroll
        for (int k = 0; k < 4; ++k) {
          bxc[k] = load_frag(wxc + (nt * 4 + k) * 512);
          bhc[k] = load_frag(whc + (nt * 4 + k) * 512);
        }
#pragma unroll
        for (int mt = 0; mt < 4; ++mt) {
          v8f acc = zero_c();
#pragma unroll
          for (int k = 0; k < 4; ++k)
            acc = wmma_f16(load_frag(inp + (mt * 4 + k) * 512), bxc[k], acc);
#pragma unroll
          for (int k = 0; k < 4; ++k)
            acc = wmma_f16(load_frag(rh + (mt * 4 + k) * 512), bhc[k], acc);
          store_frag_c(pre + mt * 16 * U_ + nt * 16, U_, acc);
        }
      }
      __syncthreads();

#if HAVE_TDM
      // last consumer of inp is done: overlap next x_t DMA with LN2 phase
      if (l == L_ - 1 && t + 1 < T_ && tid < 32)
        tdm_load_1d(0u, x + (size_t)(t + 1) * B_ * U_, B_ * U_ * 2);
#endif

      // ---- LN + tanh + state update over 128 cols; float4 scans ----
      {
        const int q = tid >> 6, r = tid & 63;
        const float4* p4 = (const float4*)(pre + r * U_ + q * 32);
        float s = 0.0f, s2 = 0.0f;
        for (int j = 0; j < 8; ++j) {
          float4 v = p4[j];
          s  += v.x + v.y + v.z + v.w;
          s2 += v.x * v.x + v.y * v.y + v.z * v.z + v.w * v.w;
        }
        red[q * 64 + r] = s;
        red[256 + q * 64 + r] = s2;
        __syncthreads();
        float mean = 0.0f, msq = 0.0f;
        for (int i = 0; i < 4; ++i) {
          mean += red[i * 64 + r];
          msq  += red[256 + i * 64 + r];
        }
        mean *= (1.0f / 128.0f);
        msq  *= (1.0f / 128.0f);
        const float rs = rsqrtf(msq - mean * mean + 1e-5f);
        const float* gc = par + 1024 + l * U_;
        const float* bc = par + 1280 + l * U_;
        const int last = (l == L_ - 1);
        auto upd2 = [&](float pv, int c) {
          float val = (pv - mean) * rs * gc[c] + bc[c];
          float cc  = tanhf(val);
          float z   = (float)zb[r * U_ + c];
          const int si = a_swz64(r, c, 4);
          float hn = z * (float)hl[si] + (1.0f - z) * cc;
          hb[l * B_ * U_ + si] = (_Float16)hn;
          if (!last) {
            inp[si] = (_Float16)hn;
          } else {
            const int grow = r * T_ + t;             // global row b*T+t
            outh[(size_t)(grow >> 6) * (B_ * U_) + a_swz64(grow & 63, c, 4)] =
                (_Float16)hn;
          }
        };
        for (int j = 0; j < 8; ++j) {
          float4 v = p4[j];
          const int c0 = q * 32 + j * 4;
          upd2(v.x, c0);
          upd2(v.y, c0 + 1);
          upd2(v.z, c0 + 2);
          upd2(v.w, c0 + 3);
        }
      }
      __syncthreads();
    }

#if HAVE_TDM
    if (tid < 32) __builtin_amdgcn_s_wait_tensorcnt((short)0);
    __syncthreads();
#endif
  }
}

// ---------------- fused logits + log-softmax + NLL per 64-row block ----------
__global__ void __launch_bounds__(256)
k_logits_loss(const _Float16* __restrict__ outh,   // [256 blocks][A-swz 64x128]
              const _Float16* __restrict__ swh,    // [U][VP] B fragment-major
              const float* __restrict__ sbp,       // [VP]
              const int* __restrict__ targets,     // [B*T]
              float* __restrict__ partials) {
  __shared__ __align__(32) _Float16 As[B_ * U_];
  __shared__ float tile[B_ * 128];
  __shared__ float rmax[B_], rsum[B_], rtgt[B_];
  __shared__ float red_m[4][B_], red_s[4][B_];
  __shared__ float nll[B_];

  const int tid  = threadIdx.x;
  const int wave = tid >> 5;
  const int m0   = blockIdx.x * B_;

  {
    const uint4* s = (const uint4*)(outh + (size_t)blockIdx.x * B_ * U_);
    uint4* d = (uint4*)As;
    for (int i = tid; i < (B_ * U_) / 8; i += 256) d[i] = s[i];
  }
  if (tid < B_) { rmax[tid] = -3.0e38f; rsum[tid] = 0.0f; rtgt[tid] = 0.0f; }

  // pipeline: B fragments for strip 0
  v16h bw[4], bn[4];
#pragma unroll
  for (int k = 0; k < 4; ++k) bw[k] = load_frag(swh + (wave * 4 + k) * 512);
  __syncthreads();

  for (int n0 = 0; n0 < VP_; n0 += 128) {
#pragma unroll
    for (int mt = 0; mt < 4; ++mt) {
      v8f acc = zero_c();
#pragma unroll
      for (int k = 0; k < 4; ++k)
        acc = wmma_f16(load_frag(As + (mt * 4 + k) * 512), bw[k], acc);
      const int lane  = tid & 31;
      const int nc    = lane & 15;
      const int mbase = (lane >> 4) << 3;
      const float bias = sbp[n0 + wave * 16 + nc];
#pragma unroll
      for (int i = 0; i < 8; ++i)
        tile[(mt * 16 + mbase + i) * 128 + wave * 16 + nc] = acc[i] + bias;
    }
    // prefetch next strip's B fragments; overlaps with the reduction phase
    if (n0 + 128 < VP_) {
      const int ntn = ((n0 + 128) >> 4) + wave;
      __builtin_prefetch(swh + ((ntn + 8) * 4) * 512, 0, 1);
#pragma unroll
      for (int k = 0; k < 4; ++k) bn[k] = load_frag(swh + (ntn * 4 + k) * 512);
    } else {
#pragma unroll
      for (int k = 0; k < 4; ++k) bn[k] = bw[k];
    }
    __syncthreads();

    {  // online logsumexp over this 128-col strip; float4 scans, 4 thr/row
      const int q = tid >> 6, r = tid & 63;
      const float4* t4 = (const float4*)(tile + r * 128 + q * 32);
      float mx = -3.0e38f, sm = 0.0f;
      auto upd = [&](float v) {
        if (v > mx) { sm = sm * __expf(mx - v) + 1.0f; mx = v; }
        else        { sm += __expf(v - mx); }
      };
      for (int j = 0; j < 8; ++j) {
        float4 v4 = t4[j];
        upd(v4.x); upd(v4.y); upd(v4.z); upd(v4.w);
      }
      red_m[q][r] = mx;
      red_s[q][r] = sm;
    }
    __syncthreads();
    if (tid < B_) {
      const int r = tid;
      float M = red_m[0][r];
      for (int i = 1; i < 4; ++i) M = fmaxf(M, red_m[i][r]);
      float S = 0.0f;
      for (int i = 0; i < 4; ++i) S += red_s[i][r] * __expf(red_m[i][r] - M);
      float Rm = rmax[r], Rs = rsum[r];
      float nm = fmaxf(Rm, M);
      rsum[r] = Rs * __expf(Rm - nm) + S * __expf(M - nm);
      rmax[r] = nm;
      const int tc = targets[m0 + r];
      if (tc >= n0 && tc < n0 + 128) rtgt[r] = tile[r * 128 + (tc - n0)];
    }
    __syncthreads();
#pragma unroll
    for (int k = 0; k < 4; ++k) bw[k] = bn[k];
  }

  if (tid < B_) nll[tid] = (rmax[tid] + __logf(rsum[tid])) - rtgt[tid];
  __syncthreads();
  if (tid == 0) {
    float s = 0.0f;
    for (int i = 0; i < B_; ++i) s += nll[i];
    partials[blockIdx.x] = s;
  }
}

__global__ void k_final_reduce(const float* __restrict__ partials,
                               float* __restrict__ out) {
  __shared__ float buf[256];
  const int tid = threadIdx.x;
  buf[tid] = partials[tid];
  __syncthreads();
  for (int s = 128; s > 0; s >>= 1) {
    if (tid < s) buf[tid] += buf[tid + s];
    __syncthreads();
  }
  if (tid == 0) out[0] = buf[0] * (1.0f / (float)(B_ * T_));
}

// ---------------- host orchestration ----------------

extern "C" void kernel_launch(void* const* d_in, const int* in_sizes, int n_in,
                              void* d_out, int out_size, void* d_ws, size_t ws_size,
                              hipStream_t stream) {
  const int*   inputs  = (const int*)d_in[0];
  const int*   targets = (const int*)d_in[1];
  const float* emb     = (const float*)d_in[2];
  const float* Wxg     = (const float*)d_in[3];
  const float* Whg     = (const float*)d_in[4];
  const float* g_g     = (const float*)d_in[5];
  const float* b_g     = (const float*)d_in[6];
  const float* Wxc     = (const float*)d_in[7];
  const float* Whc     = (const float*)d_in[8];
  const float* g_c     = (const float*)d_in[9];
  const float* b_c     = (const float*)d_in[10];
  const float* sw      = (const float*)d_in[11];
  const float* sb      = (const float*)d_in[12];

  char* ws = (char*)d_ws;
  size_t o = 0;
  auto carve = [&](size_t bytes) {
    size_t cur = o;
    o = (o + bytes + 255) & ~(size_t)255;
    return cur;
  };
  _Float16* x_h   = (_Float16*)(ws + carve((size_t)T_ * B_ * U_ * 2));
  _Float16* wxg_h = (_Float16*)(ws + carve((size_t)L_ * U_ * 2 * U_ * 2));
  _Float16* whg_h = (_Float16*)(ws + carve((size_t)L_ * U_ * 2 * U_ * 2));
  _Float16* wxc_h = (_Float16*)(ws + carve((size_t)L_ * U_ * U_ * 2));
  _Float16* whc_h = (_Float16*)(ws + carve((size_t)L_ * U_ * U_ * 2));
  _Float16* sw_h  = (_Float16*)(ws + carve((size_t)U_ * VP_ * 2));
  float*    sb_p  = (float*)(ws + carve((size_t)VP_ * 4));
  _Float16* out_h = (_Float16*)(ws + carve((size_t)B_ * T_ * U_ * 2));
  float*    parts = (float*)(ws + carve(256 * 4));

  const int TPB = 256;
  for (int l = 0; l < L_; ++l) {
    k_swz_w<<<(U_ * 2 * U_ + TPB - 1) / TPB, TPB, 0, stream>>>(
        Wxg + l * U_ * 2 * U_, wxg_h + l * U_ * 2 * U_, U_, 2 * U_);
    k_swz_w<<<(U_ * 2 * U_ + TPB - 1) / TPB, TPB, 0, stream>>>(
        Whg + l * U_ * 2 * U_, whg_h + l * U_ * 2 * U_, U_, 2 * U_);
    k_swz_w<<<(U_ * U_ + TPB - 1) / TPB, TPB, 0, stream>>>(
        Wxc + l * U_ * U_, wxc_h + l * U_ * U_, U_, U_);
    k_swz_w<<<(U_ * U_ + TPB - 1) / TPB, TPB, 0, stream>>>(
        Whc + l * U_ * U_, whc_h + l * U_ * U_, U_, U_);
  }
  k_swz_sw<<<(U_ * VP_ + TPB - 1) / TPB, TPB, 0, stream>>>(sw, sw_h);
  k_pad_sb<<<(VP_ + TPB - 1) / TPB, TPB, 0, stream>>>(sb, sb_p);
  k_embed<<<(T_ * B_ * U_ + TPB - 1) / TPB, TPB, 0, stream>>>(emb, inputs, x_h);

  const int GRU_LDS = B_ * U_ * 2      /* inp */
                    + L_ * B_ * U_ * 2 /* hb  */
                    + B_ * U_ * 2      /* rh  */
                    + B_ * U_ * 2      /* zb  */
                    + B_ * 2 * U_ * 4  /* pre */
                    + 512 * 4          /* red */
                    + 1536 * 4;        /* par */
  k_gru<<<1, 256, GRU_LDS, stream>>>(x_h, wxg_h, whg_h, wxc_h, whc_h,
                                     g_g, b_g, g_c, b_c, out_h);

  k_logits_loss<<<(B_ * T_) / B_, 256, 0, stream>>>(out_h, sw_h, sb_p, targets, parts);
  k_final_reduce<<<1, 256, 0, stream>>>(parts, (float*)d_out);
}
